// GNN_62483184222883
// MI455X (gfx1250) — matrix operand order repaired and checked by reading, not compiled, not run on hardware
//
#include <hip/hip_runtime.h>
#include <hip/hip_bf16.h>

// ---------------------------------------------------------------------------
// GIN forward for MI455X (gfx1250, wave32).
// GEMMs: v_wmma_f32_16x16x32_bf16, LDS tiles filled via async global->LDS
// copies (ASYNCcnt) with double buffering when the toolchain exposes them.
// ---------------------------------------------------------------------------

#define NNODES   100000
#define NEDGES   200000
#define EMB      300
#define NGRAPHS  5000
#define NLAYER   5
#define K1PAD    320      // 300 padded to 32 (K of GEMM1, also h2 row stride)
#define N1PAD    640      // 600 padded to 64 (N of GEMM1 = K of GEMM2)
#define N2PAD    320      // 300 padded to 64 (N of GEMM2)
#define LDA      40       // LDS row stride in bf16: 32 data + 8 pad (16B align)
#define LDB      40

typedef __bf16 bf16;
typedef __attribute__((ext_vector_type(16))) __bf16 v16bf;
typedef __attribute__((ext_vector_type(8)))  __bf16 v8bf;
typedef __attribute__((ext_vector_type(8)))  float  v8f;
typedef __attribute__((ext_vector_type(4)))  int    v4i;

#if __has_builtin(__builtin_amdgcn_global_load_async_to_lds_b128)
#define HAVE_ASYNC 1
#define NBUF 2
static __device__ __forceinline__ void async_cp16(const bf16* g, bf16* l) {
  __builtin_amdgcn_global_load_async_to_lds_b128(
      (__attribute__((address_space(1))) v4i*)g,
      (__attribute__((address_space(3))) v4i*)l, 0, 0);
}
#else
#define HAVE_ASYNC 0
#define NBUF 1
#endif

#if __has_builtin(__builtin_amdgcn_s_wait_asynccnt)
#define WAIT_ASYNC(n) __builtin_amdgcn_s_wait_asynccnt(n)
#else
#define WAIT_ASYNC(n) asm volatile("s_wait_asynccnt %0" :: "i"(n) : "memory")
#endif

static __device__ __forceinline__ v16bf ld_frag(const bf16* p0, const bf16* p1) {
  union { v16bf v; v8bf h[2]; } u;
  u.h[0] = *(const v8bf*)p0;
  u.h[1] = *(const v8bf*)p1;
  return u.v;
}

// ---------------- elementwise / setup kernels ------------------------------

__global__ void zero_f32(float* p, int n) {
  int i = blockIdx.x * blockDim.x + threadIdx.x;
  if (i < n) p[i] = 0.0f;
}

__global__ void init_h(const int* __restrict__ x,
                       const float* __restrict__ xe1,
                       const float* __restrict__ xe2,
                       float* __restrict__ h) {
  int i = blockIdx.x;
  int c = threadIdx.x;
  if (c < EMB) {
    int a = x[2 * i], b = x[2 * i + 1];
    h[(size_t)i * EMB + c] = xe1[(size_t)a * EMB + c] + xe2[(size_t)b * EMB + c];
  }
}

// agg[i] = h[i] + eemb(self-loop types 0,0); pad cols [300,320) with zeros.
__global__ void agg_init(const float* __restrict__ h,
                         const float* __restrict__ e1,
                         const float* __restrict__ e2,
                         float* __restrict__ agg) {
  int i = blockIdx.x;
  int c = threadIdx.x;
  if (c < K1PAD) {
    float v = 0.0f;
    if (c < EMB) v = h[(size_t)i * EMB + c] + e1[c] + e2[c];
    agg[(size_t)i * K1PAD + c] = v;
  }
}

// one block per edge: agg[dst] += h[src] + e1[type1] + e2[type2]
__global__ void edge_scatter(const int* __restrict__ ei,   // [2][E]
                             const int* __restrict__ ea,   // [E][2]
                             const float* __restrict__ h,
                             const float* __restrict__ e1,
                             const float* __restrict__ e2,
                             float* __restrict__ agg) {
  int e = blockIdx.x;
  int s  = ei[e];
  int d  = ei[NEDGES + e];
  int t1 = ea[2 * e];
  int t2 = ea[2 * e + 1];
  int c = threadIdx.x;
  if (c < EMB) {
    float m = h[(size_t)s * EMB + c] + e1[(size_t)t1 * EMB + c] + e2[(size_t)t2 * EMB + c];
    atomicAdd(&agg[(size_t)d * K1PAD + c], m);
  }
}

// agg f32 [N,320] -> bf16 [N,320] (4 elems/thread)
__global__ void cvt_agg(const float* __restrict__ a, bf16* __restrict__ ab, int n4) {
  int i = blockIdx.x * blockDim.x + threadIdx.x;
  if (i < n4) {
    float4 v = *(const float4*)(a + (size_t)i * 4);
    union { uint2 u; bf16 b[4]; } t;
    t.b[0] = (bf16)v.x; t.b[1] = (bf16)v.y; t.b[2] = (bf16)v.z; t.b[3] = (bf16)v.w;
    *(uint2*)(ab + (size_t)i * 4) = t.u;
  }
}

// W1 [300][600] f32 -> k-major bf16 [640][320]  (Wb[n*320+k])
__global__ void cvt_w1(const float* __restrict__ W, bf16* __restrict__ Wb) {
  int idx = blockIdx.x * blockDim.x + threadIdx.x;
  if (idx < N1PAD * K1PAD) {
    int n = idx / K1PAD, k = idx % K1PAD;
    float v = (k < EMB && n < 2 * EMB) ? W[(size_t)k * (2 * EMB) + n] : 0.0f;
    Wb[idx] = (bf16)v;
  }
}

// W2 [600][300] f32 -> k-major bf16 [320][640]  (Wb[n*640+k])
__global__ void cvt_w2(const float* __restrict__ W, bf16* __restrict__ Wb) {
  int idx = blockIdx.x * blockDim.x + threadIdx.x;
  if (idx < N2PAD * N1PAD) {
    int n = idx / N1PAD, k = idx % N1PAD;
    float v = (k < 2 * EMB && n < EMB) ? W[(size_t)k * EMB + n] : 0.0f;
    Wb[idx] = (bf16)v;
  }
}

// ---------------- unified WMMA GEMM ----------------------------------------
// C[M, *] = A[M, KTOT] x Bt[NPAD, KTOT]^T ; block tile 128x64, 8 waves (4x2),
// each wave owns 2x2 16x16 WMMA tiles. A and Bt are k-major bf16.

template <int KTOT, bool RELU_BF16_OUT>
__global__ __launch_bounds__(256)
void gemm_wmma(const bf16* __restrict__ A,
               const bf16* __restrict__ Bt,
               const float* __restrict__ bias, int validN,
               bf16* __restrict__ Cb, float* __restrict__ Cf,
               int cstride, int M) {
  __shared__ __align__(16) bf16 sA[NBUF][128 * LDA];
  __shared__ __align__(16) bf16 sB[NBUF][64 * LDB];
  const int tid  = threadIdx.x;
  const int lane = tid & 31;
  const int wave = tid >> 5;
  const int wm = wave & 3, wn = wave >> 2;
  const int half = lane >> 4, l16 = lane & 15;
  const int m0 = blockIdx.x * 128;
  const int n0 = blockIdx.y * 64;
  const int KSTEPS = KTOT / 32;

  v8f acc[2][2] = {};

#if HAVE_ASYNC
  // 3 async b128 copies per thread per tile: A 128x32 (2/thread), B 64x32 (1/thread)
  auto issue = [&](int kk, int buf) {
    const int k0 = kk * 32;
    #pragma unroll
    for (int p = 0; p < 2; ++p) {
      int cid = tid + p * 256;
      int row = cid >> 2, c8 = cid & 3;
      int gr = m0 + row; if (gr >= M) gr = M - 1;      // clamp: rows >= M discarded
      async_cp16(A + (size_t)gr * KTOT + k0 + c8 * 8, &sA[buf][row * LDA + c8 * 8]);
    }
    {
      int row = tid >> 2, c8 = tid & 3;
      async_cp16(Bt + (size_t)(n0 + row) * KTOT + k0 + c8 * 8, &sB[buf][row * LDB + c8 * 8]);
    }
  };
  issue(0, 0);
  if (KSTEPS > 1) issue(1, 1);
#endif

  for (int kk = 0; kk < KSTEPS; ++kk) {
#if HAVE_ASYNC
    if (kk + 1 < KSTEPS) WAIT_ASYNC(3); else WAIT_ASYNC(0);  // tile kk landed
    __syncthreads();
    const bf16* sa = &sA[kk & 1][0];
    const bf16* sb = &sB[kk & 1][0];
#else
    const int k0 = kk * 32;
    __syncthreads();
    #pragma unroll
    for (int p = 0; p < 2; ++p) {
      int cid = tid + p * 256;
      int row = cid >> 2, c8 = cid & 3;
      int gr = m0 + row; if (gr >= M) gr = M - 1;
      *(uint4*)&sA[0][row * LDA + c8 * 8] =
          *(const uint4*)(A + (size_t)gr * KTOT + k0 + c8 * 8);
    }
    {
      int row = tid >> 2, c8 = tid & 3;
      *(uint4*)&sB[0][row * LDB + c8 * 8] =
          *(const uint4*)(Bt + (size_t)(n0 + row) * KTOT + k0 + c8 * 8);
    }
    __syncthreads();
    const bf16* sa = &sA[0][0];
    const bf16* sb = &sB[0][0];
#endif

    v16bf aF[2], bF[2];
    #pragma unroll
    for (int mi = 0; mi < 2; ++mi) {
      const bf16* r = &sa[(wm * 32 + mi * 16 + l16) * LDA + 8 * half];
      aF[mi] = ld_frag(r, r + 16);        // K = 8*half+(0..7), 16+8*half+(0..7)
    }
    #pragma unroll
    for (int ni = 0; ni < 2; ++ni) {
      const bf16* r = &sb[(wn * 32 + ni * 16 + l16) * LDB + 8 * half];
      bF[ni] = ld_frag(r, r + 16);        // K = 8*half+(0..7), 16+8*half+(0..7)
    }
    #pragma unroll
    for (int mi = 0; mi < 2; ++mi)
      #pragma unroll
      for (int ni = 0; ni < 2; ++ni)
        acc[mi][ni] = __builtin_amdgcn_wmma_f32_16x16x32_bf16(
            false, aF[mi], false, bF[ni], (short)0, acc[mi][ni], false, false);

#if HAVE_ASYNC
    __syncthreads();                       // everyone done reading buf kk&1
    if (kk + 2 < KSTEPS) issue(kk + 2, kk & 1);
#endif
  }

  #pragma unroll
  for (int mi = 0; mi < 2; ++mi)
    #pragma unroll
    for (int ni = 0; ni < 2; ++ni) {
      int col = n0 + wn * 32 + ni * 16 + l16;
      if (RELU_BF16_OUT) {
        float bv = (col < validN) ? bias[col] : 0.0f;   // padded cols -> 0
        #pragma unroll
        for (int r = 0; r < 8; ++r) {
          int row = m0 + wm * 32 + mi * 16 + 8 * half + r;
          if (row < M) {
            float v = acc[mi][ni][r] + bv;
            v = v > 0.f ? v : 0.f;
            Cb[(size_t)row * cstride + col] = (bf16)v;
          }
        }
      } else {
        if (col < validN) {
          float bv = bias[col];
          #pragma unroll
          for (int r = 0; r < 8; ++r) {
            int row = m0 + wm * 32 + mi * 16 + 8 * half + r;
            if (row < M)
              Cf[(size_t)row * cstride + col] = acc[mi][ni][r] + bv;
          }
        }
      }
    }
}

// ---------------- batch norm ----------------------------------------------

#define BN_SEGS 64

__global__ void bn_partial(const float* __restrict__ h2, float* __restrict__ stat, int M) {
  int c = blockIdx.x;
  int seg = blockIdx.y;
  int chunk = (M + BN_SEGS - 1) / BN_SEGS;
  int i0 = seg * chunk;
  int i1 = i0 + chunk; if (i1 > M) i1 = M;
  float s = 0.f, q = 0.f;
  for (int i = i0 + threadIdx.x; i < i1; i += 256) {
    float v = h2[(size_t)i * K1PAD + c];
    s += v; q += v * v;
  }
  __shared__ float ss[256], sq[256];
  ss[threadIdx.x] = s; sq[threadIdx.x] = q;
  __syncthreads();
  for (int o = 128; o > 0; o >>= 1) {
    if (threadIdx.x < o) {
      ss[threadIdx.x] += ss[threadIdx.x + o];
      sq[threadIdx.x] += sq[threadIdx.x + o];
    }
    __syncthreads();
  }
  if (threadIdx.x == 0) {
    atomicAdd(&stat[c], ss[0]);
    atomicAdd(&stat[EMB + c], sq[0]);
  }
}

__global__ void bn_final(float* stat, int M) {
  int c = threadIdx.x;
  if (c < EMB) {
    float inv_n = 1.0f / (float)M;
    float mean = stat[c] * inv_n;
    float var  = stat[EMB + c] * inv_n - mean * mean;
    stat[2 * EMB + c] = mean;
    stat[3 * EMB + c] = rsqrtf(var + 1e-5f);
  }
}

__global__ void bn_apply(const float* __restrict__ h2,
                         const float* __restrict__ stat,
                         const float* __restrict__ gamma,
                         const float* __restrict__ beta,
                         float* __restrict__ h, int do_relu) {
  int i = blockIdx.x;
  int c = threadIdx.x;
  if (c < EMB) {
    float v = (h2[(size_t)i * K1PAD + c] - stat[2 * EMB + c]) * stat[3 * EMB + c]
              * gamma[c] + beta[c];
    if (do_relu) v = v > 0.f ? v : 0.f;
    h[(size_t)i * EMB + c] = v;
  }
}

// ---------------- pooling --------------------------------------------------

__global__ void pool_scatter(const float* __restrict__ h,
                             const int* __restrict__ batch,
                             float* __restrict__ pool,
                             float* __restrict__ cnt) {
  int i = blockIdx.x;
  int g = batch[i];
  int c = threadIdx.x;
  if (c < EMB) atomicAdd(&pool[(size_t)g * EMB + c], h[(size_t)i * EMB + c]);
  if (c == 0)  atomicAdd(&cnt[g], 1.0f);
}

__global__ void pool_final(const float* __restrict__ pool,
                           const float* __restrict__ cnt,
                           float* __restrict__ out) {
  int idx = blockIdx.x * blockDim.x + threadIdx.x;
  if (idx < NGRAPHS * EMB) {
    int g = idx / EMB;
    float n = cnt[g];
    out[idx] = pool[idx] / (n > 1.0f ? n : 1.0f);
  }
}

// ---------------- host orchestration ---------------------------------------

extern "C" void kernel_launch(void* const* d_in, const int* in_sizes, int n_in,
                              void* d_out, int out_size, void* d_ws, size_t ws_size,
                              hipStream_t stream) {
  (void)in_sizes; (void)n_in; (void)out_size; (void)ws_size;
  const int*   x     = (const int*)  d_in[0];
  const int*   ei    = (const int*)  d_in[1];
  const int*   ea    = (const int*)  d_in[2];
  const int*   batch = (const int*)  d_in[3];
  const float* xe1   = (const float*)d_in[4];
  const float* xe2   = (const float*)d_in[5];
  const float* ee1   = (const float*)d_in[6];   // [5][6][300]
  const float* ee2   = (const float*)d_in[7];   // [5][3][300]
  const float* W1    = (const float*)d_in[8];   // [5][300][600]
  const float* b1    = (const float*)d_in[9];   // [5][600]
  const float* W2    = (const float*)d_in[10];  // [5][600][300]
  const float* b2    = (const float*)d_in[11];  // [5][300]
  const float* gamma = (const float*)d_in[12];  // [5][300]
  const float* beta  = (const float*)d_in[13];  // [5][300]
  float* out = (float*)d_out;

  char* ws = (char*)d_ws;
  size_t off = 0;
  float* h     = (float*)(ws + off); off += (size_t)NNODES * EMB   * 4;  // 120 MB
  float* agg   = (float*)(ws + off); off += (size_t)NNODES * K1PAD * 4;  // 128 MB (h2 aliases)
  bf16*  aggb  = (bf16*) (ws + off); off += (size_t)NNODES * K1PAD * 2;  //  64 MB
  bf16*  a2    = (bf16*) (ws + off); off += (size_t)NNODES * N1PAD * 2;  // 128 MB
  bf16*  w1b   = (bf16*) (ws + off); off += (size_t)N1PAD * K1PAD * 2;   // k-major [640][320]
  bf16*  w2b   = (bf16*) (ws + off); off += (size_t)N2PAD * N1PAD * 2;   // k-major [320][640]
  off = (off + 255) & ~(size_t)255;
  float* stat  = (float*)(ws + off); off += 4 * EMB * 4; off = (off + 255) & ~(size_t)255;
  float* pool  = (float*)(ws + off); off += (size_t)NGRAPHS * EMB * 4;
  float* cnt   = (float*)(ws + off); off += (size_t)NGRAPHS * 4;
  float* h2    = agg;   // GEMM2 output reuses agg f32 (dead after cvt_agg)

  const dim3 g1((NNODES + 127) / 128, N1PAD / 64);   // 782 x 10
  const dim3 g2((NNODES + 127) / 128, N2PAD / 64);   // 782 x 5
  const int  n4 = NNODES * K1PAD / 4;

  init_h<<<NNODES, 320, 0, stream>>>(x, xe1, xe2, h);

  for (int l = 0; l < NLAYER; ++l) {
    const float* e1l = ee1 + (size_t)l * 6 * EMB;
    const float* e2l = ee2 + (size_t)l * 3 * EMB;

    cvt_w1<<<(N1PAD * K1PAD + 255) / 256, 256, 0, stream>>>(W1 + (size_t)l * EMB * 2 * EMB, w1b);
    cvt_w2<<<(N2PAD * N1PAD + 255) / 256, 256, 0, stream>>>(W2 + (size_t)l * 2 * EMB * EMB, w2b);

    agg_init<<<NNODES, 320, 0, stream>>>(h, e1l, e2l, agg);
    edge_scatter<<<NEDGES, 320, 0, stream>>>(ei, ea, h, e1l, e2l, agg);
    cvt_agg<<<(n4 + 255) / 256, 256, 0, stream>>>(agg, aggb, n4);

    gemm_wmma<K1PAD, true ><<<g1, 256, 0, stream>>>(aggb, w1b, b1 + (size_t)l * 2 * EMB,
                                                    2 * EMB, a2, nullptr, N1PAD, NNODES);
    gemm_wmma<N1PAD, false><<<g2, 256, 0, stream>>>(a2, w2b, b2 + (size_t)l * EMB,
                                                    EMB, nullptr, h2, K1PAD, NNODES);

    zero_f32<<<(2 * EMB + 255) / 256, 256, 0, stream>>>(stat, 2 * EMB);
    bn_partial<<<dim3(EMB, BN_SEGS), 256, 0, stream>>>(h2, stat, NNODES);
    bn_final<<<1, 320, 0, stream>>>(stat, NNODES);
    bn_apply<<<NNODES, 320, 0, stream>>>(h2, stat, gamma + (size_t)l * EMB,
                                         beta + (size_t)l * EMB, h, (l != NLAYER - 1));
  }

  zero_f32<<<(NGRAPHS * EMB + NGRAPHS + 255) / 256, 256, 0, stream>>>(pool, NGRAPHS * EMB + NGRAPHS);
  pool_scatter<<<NNODES, 320, 0, stream>>>(h, batch, pool, cnt);
  pool_final<<<(NGRAPHS * EMB + 255) / 256, 256, 0, stream>>>(pool, cnt, out);
}
